// QuantizedLinear_83030307766619
// MI455X (gfx1250) — matrix operand order repaired and checked by reading, not compiled, not run on hardware
//
#include <hip/hip_runtime.h>

typedef __attribute__((ext_vector_type(16))) __bf16 v16bf;
typedef __attribute__((ext_vector_type(8)))  float  v8f;

#define K_TILE     32
#define LDS_STRIDE 40   // 32 data + 8 pad bf16 per row: 80B (20 dwords, odd*4 -> bank-conflict free, 16B aligned)
#define BLK_M      256
#define BLK_N      128

union Frag {
    v16bf v;
    uint4 q[2];
};

static __device__ __forceinline__ unsigned short bf_bits(float f) {
    __bf16 h = (__bf16)f;              // RNE truncation
    return __builtin_bit_cast(unsigned short, h);
}

// ---------------------------------------------------------------------------
// Kernel 1: per-group (128 contiguous) INT2 fake-quant of weight, then split
// each dequantized value into bf16 hi + bf16 lo (w = hi + lo).
// One wave (32 lanes x 4 floats) per group.
// ---------------------------------------------------------------------------
__global__ __launch_bounds__(256)
void quant_split_kernel(const float* __restrict__ w,
                        __bf16* __restrict__ whi,
                        __bf16* __restrict__ wlo,
                        long long n_groups)
{
    const int wave = threadIdx.x >> 5;
    const int lane = threadIdx.x & 31;
    long long g = (long long)blockIdx.x * 8 + wave;
    if (g >= n_groups) return;

    const long long base = g * 128 + (long long)lane * 4;
    const float4 f = *reinterpret_cast<const float4*>(&w[base]);

    float mn = fminf(fminf(f.x, f.y), fminf(f.z, f.w));
    float mx = fmaxf(fmaxf(f.x, f.y), fmaxf(f.z, f.w));
    #pragma unroll
    for (int off = 16; off >= 1; off >>= 1) {
        mn = fminf(mn, __shfl_xor(mn, off, 32));
        mx = fmaxf(mx, __shfl_xor(mx, off, 32));
    }

    const float rng   = mx - mn;
    const bool  pass  = rng < 1e-8f;           // near-constant group: pass-through
    const float scale = rng * (1.0f / 3.0f);
    const float inv   = pass ? 0.0f : 3.0f / rng;

    float vals[4] = { f.x, f.y, f.z, f.w };
    unsigned short hi[4], lo[4];
    #pragma unroll
    for (int i = 0; i < 4; ++i) {
        float q = rintf((vals[i] - mn) * inv);  // round-half-even, matches jnp.round
        q = fminf(fmaxf(q, 0.0f), 3.0f);
        float d = pass ? vals[i] : (q * scale + mn);
        hi[i] = bf_bits(d);
        float dh = (float)__builtin_bit_cast(__bf16, hi[i]);
        lo[i] = bf_bits(d - dh);
    }

    uint2 ph, pl;
    ph.x = (unsigned)hi[0] | ((unsigned)hi[1] << 16);
    ph.y = (unsigned)hi[2] | ((unsigned)hi[3] << 16);
    pl.x = (unsigned)lo[0] | ((unsigned)lo[1] << 16);
    pl.y = (unsigned)lo[2] | ((unsigned)lo[3] << 16);
    *reinterpret_cast<uint2*>(&whi[base]) = ph;
    *reinterpret_cast<uint2*>(&wlo[base]) = pl;
}

// ---------------------------------------------------------------------------
// GEMM pipeline stage: registers holding the NEXT K-chunk's global data.
// ---------------------------------------------------------------------------
struct Stage {
    float4 a[8];        // 256 rows x 32 f32 of x, 256 threads -> 8 float4 each
    uint4  bh[2];       // 128 rows x 32 bf16 of w_hi -> 2 uint4 each
    uint4  bl[2];       // 128 rows x 32 bf16 of w_lo
};

static __device__ __forceinline__ void stage_load(Stage& s,
        const float* __restrict__ x,
        const __bf16* __restrict__ whi, const __bf16* __restrict__ wlo,
        int tid, int m0, int n0, int K, int k0)
{
    #pragma unroll
    for (int p = 0; p < 8; ++p) {
        int linear = p * 256 + tid;
        int row = linear >> 3, q4 = linear & 7;
        s.a[p] = *reinterpret_cast<const float4*>(&x[(size_t)(m0 + row) * K + k0 + q4 * 4]);
    }
    #pragma unroll
    for (int p = 0; p < 2; ++p) {
        int linear = p * 256 + tid;
        int row = linear >> 2, j = linear & 3;
        size_t src = (size_t)(n0 + row) * K + k0 + j * 8;
        s.bh[p] = *reinterpret_cast<const uint4*>(&whi[src]);
        s.bl[p] = *reinterpret_cast<const uint4*>(&wlo[src]);
    }
}

static __device__ __forceinline__ void stage_store(const Stage& s,
        __bf16* a_hi, __bf16* a_lo, __bf16* b_hi, __bf16* b_lo, int tid)
{
    #pragma unroll
    for (int p = 0; p < 8; ++p) {
        int linear = p * 256 + tid;
        int row = linear >> 3, q4 = linear & 7;
        float v[4] = { s.a[p].x, s.a[p].y, s.a[p].z, s.a[p].w };
        unsigned short hb[4], lb[4];
        #pragma unroll
        for (int i = 0; i < 4; ++i) {
            hb[i] = bf_bits(v[i]);
            float fh = (float)__builtin_bit_cast(__bf16, hb[i]);
            lb[i] = bf_bits(v[i] - fh);
        }
        uint2 ph, pl;
        ph.x = (unsigned)hb[0] | ((unsigned)hb[1] << 16);
        ph.y = (unsigned)hb[2] | ((unsigned)hb[3] << 16);
        pl.x = (unsigned)lb[0] | ((unsigned)lb[1] << 16);
        pl.y = (unsigned)lb[2] | ((unsigned)lb[3] << 16);
        *reinterpret_cast<uint2*>(&a_hi[row * LDS_STRIDE + q4 * 4]) = ph;
        *reinterpret_cast<uint2*>(&a_lo[row * LDS_STRIDE + q4 * 4]) = pl;
    }
    #pragma unroll
    for (int p = 0; p < 2; ++p) {
        int linear = p * 256 + tid;
        int row = linear >> 2, j = linear & 3;
        *reinterpret_cast<uint4*>(&b_hi[row * LDS_STRIDE + j * 8]) = s.bh[p];
        *reinterpret_cast<uint4*>(&b_lo[row * LDS_STRIDE + j * 8]) = s.bl[p];
    }
}

// ---------------------------------------------------------------------------
// Kernel 2: out[m,n] = sum_k x[m,k] * wq[n,k] + bias[n]
// 256x128 tile per block (8 waves, 4(M) x 2(N) wave grid, 64x64 per wave).
// Register-staged software pipeline: global loads for chunk k+1 issue before
// the WMMA burst of chunk k. 3-product split-bf16 WMMA for near-fp32 accuracy.
// ---------------------------------------------------------------------------
__global__ __launch_bounds__(256)
void gemm_bf16split_kernel(const float*  __restrict__ x,
                           const __bf16* __restrict__ whi,
                           const __bf16* __restrict__ wlo,
                           const float*  __restrict__ bias,
                           float*        __restrict__ out,
                           int M, int N, int K)
{
    __shared__ alignas(16) __bf16 a_hi[BLK_M * LDS_STRIDE];
    __shared__ alignas(16) __bf16 a_lo[BLK_M * LDS_STRIDE];
    __shared__ alignas(16) __bf16 b_hi[BLK_N * LDS_STRIDE];
    __shared__ alignas(16) __bf16 b_lo[BLK_N * LDS_STRIDE];

    const int tid  = threadIdx.x;
    const int wave = tid >> 5;
    const int lane = tid & 31;
    const int hh   = lane >> 4;     // half-wave select
    const int ln16 = lane & 15;

    const int wm = wave & 3;        // wave's 64-row slab in M
    const int wn = wave >> 2;       // wave's 64-col slab in N

    const int block_n0 = blockIdx.x * BLK_N;
    const int block_m0 = blockIdx.y * BLK_M;

    v8f acc[4][4];
    #pragma unroll
    for (int mi = 0; mi < 4; ++mi)
        #pragma unroll
        for (int ni = 0; ni < 4; ++ni)
            acc[mi][ni] = (v8f){0.f, 0.f, 0.f, 0.f, 0.f, 0.f, 0.f, 0.f};

    Stage s;
    stage_load(s, x, whi, wlo, tid, block_m0, block_n0, K, 0);
    stage_store(s, a_hi, a_lo, b_hi, b_lo, tid);
    __syncthreads();

    for (int k0 = 0; k0 < K; k0 += K_TILE) {
        const bool more = (k0 + K_TILE) < K;

        // ---- A fragments (held in VGPRs across the ni loop) ----
        Frag afh[4], afl[4];
        #pragma unroll
        for (int mi = 0; mi < 4; ++mi) {
            int m = wm * 64 + mi * 16 + ln16;
            afh[mi].q[0] = *reinterpret_cast<const uint4*>(&a_hi[m * LDS_STRIDE + hh * 8]);
            afh[mi].q[1] = *reinterpret_cast<const uint4*>(&a_hi[m * LDS_STRIDE + 16 + hh * 8]);
            afl[mi].q[0] = *reinterpret_cast<const uint4*>(&a_lo[m * LDS_STRIDE + hh * 8]);
            afl[mi].q[1] = *reinterpret_cast<const uint4*>(&a_lo[m * LDS_STRIDE + 16 + hh * 8]);
        }

        // ---- software pipeline: issue next chunk's global loads now ----
        if (more)
            stage_load(s, x, whi, wlo, tid, block_m0, block_n0, K, k0 + K_TILE);
        if (k0 + 2 * K_TILE < K) {  // warm L2 one more chunk ahead
            __builtin_prefetch(&x[(size_t)(block_m0 + tid) * K + k0 + 2 * K_TILE], 0, 0);
            const __bf16* wp = (tid < 128) ? whi : wlo;
            __builtin_prefetch(&wp[(size_t)(block_n0 + (tid & 127)) * K + k0 + 2 * K_TILE], 0, 0);
        }

        // ---- 48 WMMAs: B fragments streamed per-ni (2 live at a time) ----
        #pragma unroll
        for (int ni = 0; ni < 4; ++ni) {
            int n = wn * 64 + ni * 16 + ln16;
            Frag bh, bl;
            bh.q[0] = *reinterpret_cast<const uint4*>(&b_hi[n * LDS_STRIDE + hh * 8]);
            bh.q[1] = *reinterpret_cast<const uint4*>(&b_hi[n * LDS_STRIDE + 16 + hh * 8]);
            bl.q[0] = *reinterpret_cast<const uint4*>(&b_lo[n * LDS_STRIDE + hh * 8]);
            bl.q[1] = *reinterpret_cast<const uint4*>(&b_lo[n * LDS_STRIDE + 16 + hh * 8]);
            #pragma unroll
            for (int mi = 0; mi < 4; ++mi) {
                acc[mi][ni] = __builtin_amdgcn_wmma_f32_16x16x32_bf16(
                    false, afh[mi].v, false, bh.v, (short)0, acc[mi][ni], false, false);
                acc[mi][ni] = __builtin_amdgcn_wmma_f32_16x16x32_bf16(
                    false, afh[mi].v, false, bl.v, (short)0, acc[mi][ni], false, false);
                acc[mi][ni] = __builtin_amdgcn_wmma_f32_16x16x32_bf16(
                    false, afl[mi].v, false, bh.v, (short)0, acc[mi][ni], false, false);
            }
        }

        if (more) {
            __syncthreads();
            stage_store(s, a_hi, a_lo, b_hi, b_lo, tid);
            __syncthreads();
        }
    }

    // ---- epilogue: D layout -> m = r + 8*half, n = lane%16 ----
    #pragma unroll
    for (int ni = 0; ni < 4; ++ni) {
        int n = block_n0 + wn * 64 + ni * 16 + ln16;
        float bv = bias[n];
        #pragma unroll
        for (int mi = 0; mi < 4; ++mi) {
            #pragma unroll
            for (int r = 0; r < 8; ++r) {
                int m = block_m0 + wm * 64 + mi * 16 + hh * 8 + r;
                out[(size_t)m * N + n] = acc[mi][ni][r] + bv;
            }
        }
    }
}

extern "C" void kernel_launch(void* const* d_in, const int* in_sizes, int n_in,
                              void* d_out, int out_size, void* d_ws, size_t ws_size,
                              hipStream_t stream) {
    const float* x    = (const float*)d_in[0];
    const float* w    = (const float*)d_in[1];
    const float* bias = (const float*)d_in[2];
    float* out = (float*)d_out;

    const long long N = in_sizes[2];                       // 4096 (bias length = D_OUT)
    const long long K = (long long)in_sizes[1] / N;        // 4096
    const long long M = (long long)in_sizes[0] / K;        // 8192

    __bf16* whi = (__bf16*)d_ws;
    __bf16* wlo = whi + (size_t)(N * K);

    const long long n_groups = (N * K) / 128;
    const int qblocks = (int)((n_groups + 7) / 8);
    quant_split_kernel<<<qblocks, 256, 0, stream>>>(w, whi, wlo, n_groups);

    dim3 grid((unsigned)(N / BLK_N), (unsigned)(M / BLK_M));
    gemm_bf16split_kernel<<<grid, 256, 0, stream>>>(x, whi, wlo, bias, out,
                                                    (int)M, (int)N, (int)K);
}
